// Informer_52922587021391
// MI455X (gfx1250) — matrix operand order repaired and checked by reading, not compiled, not run on hardware
//
#include <hip/hip_runtime.h>
#include <math.h>

#define DM 512
#define NH 8
#define HD 64
#define BB 4

typedef __attribute__((ext_vector_type(16))) _Float16 v16h;
typedef __attribute__((ext_vector_type(8)))  float    v8f;

__device__ __forceinline__ float gelu_f(float x) {
  return 0.5f * x * (1.0f + erff(x * 0.70710678118654752f));
}

// K offset pattern inside a 16x32 f16 A fragment (contiguous: k = 16*half + 2*j, matching B)
__device__ __forceinline__ int kpatA(int half, int j) {
  return 16 * half + 2 * j;
}

// Branch-proof guarded load: clamped address, result always consumed via
// multiply (0*x not foldable without fast-math), so the compiler cannot sink
// the load into an exec-mask branch.
__device__ __forceinline__ float mload(const float* base, long idx, bool ok) {
  float v = base[ok ? idx : 0];
  return v * (ok ? 1.0f : 0.0f);
}

// ---------------------------------------------------------------------------
// Generic batched WMMA GEMM:  C = resid + act(scale * (A @ opB(W)) + bias)
// A fp32 [M,K] (lda), W fp32 (ldw; TRANSB: W[n*ldw+k]), C fp32 (ldc).
// Batch z: b=z/Hb, h=z%Hb; offsets b*s?b + h*s?h.
// Block = 128 threads (4 waves). Each wave owns a 32x64 strip of a 128x64
// tile: 2 A fragments x 4 B fragments -> 8 v_wmma_f32_16x16x32_f16 per
// 32-deep K step.
// Interior blocks: B fragments are produced cooperatively once per block
// (thread tid -> tile tid>>5, lane tid&31), staged in LDS in fragment layout,
// and read back by all 4 waves (4x reduction of B loads/converts).
// Edge blocks / K tail: clamp + mask-mul guards (no exec branching).
// ---------------------------------------------------------------------------
template <int TRANSB>
__global__ __launch_bounds__(128) void gemm_f16_wmma(
    const float* __restrict__ A, const float* __restrict__ W,
    const float* __restrict__ bias, float* __restrict__ C,
    const float* __restrict__ resid,
    int M, int N, int K, int lda, int ldw, int ldc,
    long sAb, long sAh, long sWb, long sWh, long sCb, long sCh,
    int Hb, int act, float scale)
{
  __shared__ v16h bshare[128];   // 4 tiles x 32 lanes x 32B = 4 KB

  int z  = blockIdx.z;
  int bb = z / Hb, hh = z % Hb;
  A += (long)bb * sAb + (long)hh * sAh;
  W += (long)bb * sWb + (long)hh * sWh;
  long coff = (long)bb * sCb + (long)hh * sCh;
  C += coff;
  if (resid) resid += coff;

  int wave = threadIdx.x >> 5, lane = threadIdx.x & 31;
  int half = lane >> 4, l16 = lane & 15;
  int m0 = blockIdx.y * 128 + wave * 32;
  int n0 = blockIdx.x * 64;
  v8f acc[2][4] = {};
  int row0 = m0 + l16;
  int row1 = m0 + 16 + l16;

  // Block-uniform path choice (barriers inside the fast path!)
  bool blockFull = ((blockIdx.y * 128 + 128) <= M) && ((n0 + 64) <= N);
  int Kmain = K & ~31;

  if (blockFull) {
    const float* Ar0 = A + (long)row0 * lda;
    const float* Ar1 = A + (long)row1 * lda;
    int nB = n0 + wave * 16 + l16;         // column this thread produces
    // ---------------- fast interior path (LDS-staged B) ----------------
    for (int kk = 0; kk < Kmain; kk += 32) {
      // produce this thread's B fragment (tile = wave)
      v16h bw;
      if (TRANSB) {
        const float* Wr = W + (long)nB * ldw + kk + 16 * half;
#pragma unroll
        for (int j = 0; j < 8; ++j) {
          float2 y = *(const float2*)(Wr + 2 * j);
          bw[2 * j]     = (_Float16)y.x;
          bw[2 * j + 1] = (_Float16)y.y;
        }
      } else {
        const float* Wc = W + (long)(kk + 16 * half) * ldw + nB;
#pragma unroll
        for (int j = 0; j < 8; ++j) {
          bw[2 * j]     = (_Float16)Wc[(long)(2 * j) * ldw];
          bw[2 * j + 1] = (_Float16)Wc[(long)(2 * j + 1) * ldw];
        }
      }
      __syncthreads();                 // previous iteration's reads done
      bshare[threadIdx.x] = bw;
      // A fragments (overlap with LDS store latency)
      v16h a0, a1;
#pragma unroll
      for (int j = 0; j < 8; ++j) {
        int kj = kk + kpatA(half, j);
        float2 x = *(const float2*)(Ar0 + kj);
        float2 y = *(const float2*)(Ar1 + kj);
        a0[2 * j] = (_Float16)x.x; a0[2 * j + 1] = (_Float16)x.y;
        a1[2 * j] = (_Float16)y.x; a1[2 * j + 1] = (_Float16)y.y;
      }
      __syncthreads();                 // B fragments visible
#pragma unroll
      for (int t = 0; t < 4; ++t) {
        v16h bf = bshare[t * 32 + lane];
        acc[0][t] = __builtin_amdgcn_wmma_f32_16x16x32_f16(
            false, a0, false, bf, (short)0, acc[0][t], false, false);
        acc[1][t] = __builtin_amdgcn_wmma_f32_16x16x32_f16(
            false, a1, false, bf, (short)0, acc[1][t], false, false);
      }
    }
    // K tail (mask-mul guarded; N/M in range because blockFull)
    if (Kmain < K) {
      int kk = Kmain;
      v16h a0, a1;
#pragma unroll
      for (int j = 0; j < 8; ++j) {
        int k0 = kk + kpatA(half, j);
        a0[2 * j]     = (_Float16)mload(Ar0, k0,     k0     < K);
        a0[2 * j + 1] = (_Float16)mload(Ar0, k0 + 1, k0 + 1 < K);
        a1[2 * j]     = (_Float16)mload(Ar1, k0,     k0     < K);
        a1[2 * j + 1] = (_Float16)mload(Ar1, k0 + 1, k0 + 1 < K);
      }
#pragma unroll
      for (int t = 0; t < 4; ++t) {
        int n = n0 + t * 16 + l16;
        v16h bf;
#pragma unroll
        for (int j = 0; j < 8; ++j) {
          int k0 = kk + 16 * half + 2 * j;
          if (TRANSB) {
            const float* Wr = W + (long)n * ldw;
            bf[2 * j]     = (_Float16)mload(Wr, k0,     k0     < K);
            bf[2 * j + 1] = (_Float16)mload(Wr, k0 + 1, k0 + 1 < K);
          } else {
            bf[2 * j]     = (_Float16)mload(W, (long)k0 * ldw + n,       k0     < K);
            bf[2 * j + 1] = (_Float16)mload(W, (long)(k0 + 1) * ldw + n, k0 + 1 < K);
          }
        }
        acc[0][t] = __builtin_amdgcn_wmma_f32_16x16x32_f16(
            false, a0, false, bf, (short)0, acc[0][t], false, false);
        acc[1][t] = __builtin_amdgcn_wmma_f32_16x16x32_f16(
            false, a1, false, bf, (short)0, acc[1][t], false, false);
      }
    }
  } else {
    // ---------------- guarded edge path (clamp + mask-mul) ----------------
    bool r0ok = row0 < M, r1ok = row1 < M;
    const float* Ar0 = A + (long)(r0ok ? row0 : 0) * lda;
    const float* Ar1 = A + (long)(r1ok ? row1 : 0) * lda;
    for (int kk = 0; kk < K; kk += 32) {
      v16h a0, a1;
#pragma unroll
      for (int j = 0; j < 8; ++j) {
        int k0 = kk + kpatA(half, j);
        a0[2 * j]     = (_Float16)mload(Ar0, k0,     r0ok && k0     < K);
        a0[2 * j + 1] = (_Float16)mload(Ar0, k0 + 1, r0ok && k0 + 1 < K);
        a1[2 * j]     = (_Float16)mload(Ar1, k0,     r1ok && k0     < K);
        a1[2 * j + 1] = (_Float16)mload(Ar1, k0 + 1, r1ok && k0 + 1 < K);
      }
#pragma unroll
      for (int t = 0; t < 4; ++t) {
        int n = n0 + t * 16 + l16;
        bool nok = n < N;
        const float* WrT = W + (long)(nok ? n : 0) * ldw;   // TRANSB base
        v16h bf;
#pragma unroll
        for (int j = 0; j < 8; ++j) {
          int k0 = kk + 16 * half + 2 * j;
          if (TRANSB) {
            bf[2 * j]     = (_Float16)mload(WrT, k0,     nok && k0     < K);
            bf[2 * j + 1] = (_Float16)mload(WrT, k0 + 1, nok && k0 + 1 < K);
          } else {
            long nc = nok ? n : 0;
            bf[2 * j]     = (_Float16)mload(W, (long)k0 * ldw + nc,       nok && k0     < K);
            bf[2 * j + 1] = (_Float16)mload(W, (long)(k0 + 1) * ldw + nc, nok && k0 + 1 < K);
          }
        }
        acc[0][t] = __builtin_amdgcn_wmma_f32_16x16x32_f16(
            false, a0, false, bf, (short)0, acc[0][t], false, false);
        acc[1][t] = __builtin_amdgcn_wmma_f32_16x16x32_f16(
            false, a1, false, bf, (short)0, acc[1][t], false, false);
      }
    }
  }

  // ---------------- store (C: VGPR i -> M=i + 8*half) ----------------
  if (blockFull) {
#pragma unroll
    for (int s = 0; s < 2; ++s) {
#pragma unroll
      for (int t = 0; t < 4; ++t) {
#pragma unroll
        for (int i = 0; i < 8; ++i) {
          int r = m0 + s * 16 + i + 8 * half;
          int c = n0 + t * 16 + l16;
          float v = acc[s][t][i] * scale;
          if (bias) v += bias[c];
          if (act == 1) v = gelu_f(v);
          if (resid) v += resid[(long)r * ldc + c];
          C[(long)r * ldc + c] = v;
        }
      }
    }
  } else {
#pragma unroll
    for (int s = 0; s < 2; ++s) {
#pragma unroll
      for (int t = 0; t < 4; ++t) {
#pragma unroll
        for (int i = 0; i < 8; ++i) {
          int r = m0 + s * 16 + i + 8 * half;
          int c = n0 + t * 16 + l16;
          if (r < M && c < N) {
            float v = acc[s][t][i] * scale;
            if (bias) v += bias[c];
            if (act == 1) v = gelu_f(v);
            if (resid) v += resid[(long)r * ldc + c];
            C[(long)r * ldc + c] = v;
          }
        }
      }
    }
  }
}

// ---------------------------------------------------------------------------
// LayerNorm over last dim (512). One block per row.
// ---------------------------------------------------------------------------
__global__ __launch_bounds__(256) void layernorm_k(
    const float* __restrict__ x, float* __restrict__ out,
    const float* __restrict__ g, const float* __restrict__ b)
{
  long row = blockIdx.x;
  const float* xr = x + row * DM;
  __shared__ float red[256];
  int tid = threadIdx.x;
  float s = 0.0f;
  for (int i = tid; i < DM; i += 256) s += xr[i];
  red[tid] = s; __syncthreads();
  for (int o = 128; o > 0; o >>= 1) { if (tid < o) red[tid] += red[tid + o]; __syncthreads(); }
  float mean = red[0] * (1.0f / DM);
  __syncthreads();
  float v = 0.0f;
  for (int i = tid; i < DM; i += 256) { float d = xr[i] - mean; v += d * d; }
  red[tid] = v; __syncthreads();
  for (int o = 128; o > 0; o >>= 1) { if (tid < o) red[tid] += red[tid + o]; __syncthreads(); }
  float inv = rsqrtf(red[0] * (1.0f / DM) + 1e-5f);
  for (int i = tid; i < DM; i += 256) out[row * DM + i] = (xr[i] - mean) * inv * g[i] + b[i];
}

// ---------------------------------------------------------------------------
// Embedding: circular conv (width 3, C_in=7) + temporal lin + sin/cos pos enc
// ---------------------------------------------------------------------------
__global__ __launch_bounds__(512) void embed_k(
    const float* __restrict__ x, const float* __restrict__ xm,
    const float* __restrict__ cw, const float* __restrict__ tw,
    const float* __restrict__ tb, float* __restrict__ out, int L, int Cin)
{
  int bl = blockIdx.x;
  int b = bl / L, l = bl % L;
  int d = threadIdx.x;
  float acc = tb[d];
  for (int kw = 0; kw < 3; ++kw) {
    int ls = (l + kw - 1 + L) % L;
    const float* xr = x + ((long)b * L + ls) * Cin;
    const float* wr = cw + (long)kw * Cin * DM;
    for (int c = 0; c < Cin; ++c) acc += xr[c] * wr[c * DM + d];
  }
  const float* mr = xm + ((long)b * L + l) * 4;
  for (int t = 0; t < 4; ++t) acc += mr[t] * tw[t * DM + d];
  int d2 = d & ~1;
  float dv = expf((float)d2 * (-9.210340371976184f / (float)DM));
  float ang = (float)l * dv;
  acc += (d & 1) ? cosf(ang) : sinf(ang);
  out[((long)b * L + l) * DM + d] = acc;
}

// Circular pad along L: xp[b, 0..L+2p) = x[b, (i-p) mod L]
__global__ void padcirc_k(const float* __restrict__ x, float* __restrict__ xp,
                          int L, int pad)
{
  long idx = (long)blockIdx.x * blockDim.x + threadIdx.x;
  int Lp = L + 2 * pad;
  long tot = (long)BB * Lp * DM;
  if (idx >= tot) return;
  int c = (int)(idx % DM);
  long r = idx / DM;
  int i = (int)(r % Lp);
  int b = (int)(r / Lp);
  int ls = (i - pad + L) % L;
  xp[idx] = x[((long)b * L + ls) * DM + c];
}

// BatchNorm stats over all rows, per channel
__global__ __launch_bounds__(256) void bnstats_k(
    const float* __restrict__ y, float* __restrict__ mv, int Mr)
{
  __shared__ float r1[256], r2[256];
  int c = blockIdx.x, tid = threadIdx.x;
  float s = 0.0f, s2 = 0.0f;
  for (int r = tid; r < Mr; r += 256) { float v = y[(long)r * DM + c]; s += v; s2 += v * v; }
  r1[tid] = s; r2[tid] = s2; __syncthreads();
  for (int o = 128; o > 0; o >>= 1) {
    if (tid < o) { r1[tid] += r1[tid + o]; r2[tid] += r2[tid + o]; }
    __syncthreads();
  }
  if (tid == 0) {
    float m = r1[0] / (float)Mr;
    mv[c] = m;
    mv[DM + c] = r2[0] / (float)Mr - m * m;
  }
}

__global__ void bnelu_k(float* __restrict__ y, const float* __restrict__ mv,
                        const float* __restrict__ g, const float* __restrict__ b,
                        long tot)
{
  long idx = (long)blockIdx.x * blockDim.x + threadIdx.x;
  if (idx >= tot) return;
  int c = (int)(idx % DM);
  float m = mv[c];
  float inv = rsqrtf(mv[DM + c] + 1e-5f);
  float v = (y[idx] - m) * inv * g[c] + b[c];
  y[idx] = v > 0.0f ? v : (expf(v) - 1.0f);
}

// MaxPool window 3 stride 2 with (1,1) -inf padding
__global__ void maxpool_k(const float* __restrict__ y, float* __restrict__ out,
                          int Ly, int Lo)
{
  long idx = (long)blockIdx.x * blockDim.x + threadIdx.x;
  long tot = (long)BB * Lo * DM;
  if (idx >= tot) return;
  int c = (int)(idx % DM);
  long r = idx / DM;
  int i = (int)(r % Lo);
  int b = (int)(r / Lo);
  float m = -1e30f;
  for (int t = -1; t <= 1; ++t) {
    int j = 2 * i + t;
    if (j >= 0 && j < Ly) m = fmaxf(m, y[((long)b * Ly + j) * DM + c]);
  }
  out[idx] = m;
}

// Row softmax (in place)
__global__ __launch_bounds__(128) void softmax_k(float* __restrict__ s, int cols)
{
  float* r = s + (long)blockIdx.x * cols;
  __shared__ float red[128];
  int tid = threadIdx.x;
  float mx = -1e30f;
  for (int i = tid; i < cols; i += 128) mx = fmaxf(mx, r[i]);
  red[tid] = mx; __syncthreads();
  for (int o = 64; o > 0; o >>= 1) { if (tid < o) red[tid] = fmaxf(red[tid], red[tid + o]); __syncthreads(); }
  mx = red[0]; __syncthreads();
  float sum = 0.0f;
  for (int i = tid; i < cols; i += 128) { float e = expf(r[i] - mx); r[i] = e; sum += e; }
  red[tid] = sum; __syncthreads();
  for (int o = 64; o > 0; o >>= 1) { if (tid < o) red[tid] += red[tid + o]; __syncthreads(); }
  float inv = 1.0f / red[0];
  __syncthreads();
  for (int i = tid; i < cols; i += 128) r[i] *= inv;
}

// ProbSparse: sparsity measure M(l) = max_j(q.k_samp) - mean(q.k_samp)
__global__ __launch_bounds__(128) void probM_k(
    const float* __restrict__ qb, const float* __restrict__ kb,
    float* __restrict__ Mm, int L, int Lk, int U)
{
  long idx = (long)blockIdx.x * blockDim.x + threadIdx.x;
  long tot = (long)BB * NH * L;
  if (idx >= tot) return;
  int l = (int)(idx % L);
  long bh = idx / L;
  int h = (int)(bh % NH);
  int b = (int)(bh / NH);
  const float* q = qb + ((long)b * L + l) * DM + h * HD;
  float mx = -1e30f, sm = 0.0f;
  for (int j = 0; j < U; ++j) {
    unsigned si = ((unsigned)l * 2654435761u + (unsigned)j * 40503u + 2463534242u)
                  % (unsigned)Lk;
    const float* kr = kb + ((long)b * Lk + si) * DM + h * HD;
    float d = 0.0f;
    for (int t = 0; t < HD; ++t) d += q[t] * kr[t];
    mx = fmaxf(mx, d);
    sm += d;
  }
  Mm[idx] = mx - sm / (float)Lk;
}

// Top-u indices of M per (b,h): iterative shared-memory argmax
__global__ __launch_bounds__(256) void topk_k(
    const float* __restrict__ Mm, int* __restrict__ top, int L, int u)
{
  __shared__ float mloc[512];
  __shared__ float sv[256];
  __shared__ int   si[256];
  int tid = threadIdx.x;
  const float* m = Mm + (long)blockIdx.x * L;
  for (int i = tid; i < L; i += 256) mloc[i] = m[i];
  __syncthreads();
  for (int it = 0; it < u; ++it) {
    float bv = -1e38f; int bi = 0;
    for (int i = tid; i < L; i += 256) if (mloc[i] > bv) { bv = mloc[i]; bi = i; }
    sv[tid] = bv; si[tid] = bi;
    __syncthreads();
    for (int o = 128; o > 0; o >>= 1) {
      if (tid < o && sv[tid + o] > sv[tid]) { sv[tid] = sv[tid + o]; si[tid] = si[tid + o]; }
      __syncthreads();
    }
    if (tid == 0) { top[(long)blockIdx.x * u + it] = si[0]; mloc[si[0]] = -1e38f; }
    __syncthreads();
  }
}

// ctx init: masked -> cumsum(v); else broadcast mean(v)
__global__ __launch_bounds__(64) void ctxinit_k(
    const float* __restrict__ vb, float* __restrict__ ctx,
    int L, int Lk, int maskf)
{
  int bh = blockIdx.x;
  int h = bh % NH, b = bh / NH;
  int d = threadIdx.x;
  const float* vc = vb + (long)b * Lk * DM + h * HD + d;
  float* cx = ctx + (long)bh * L * HD + d;
  if (maskf) {
    float run = 0.0f;
    for (int l = 0; l < L; ++l) { run += vc[(long)l * DM]; cx[(long)l * HD] = run; }
  } else {
    float s = 0.0f;
    for (int k = 0; k < Lk; ++k) s += vc[(long)k * DM];
    s /= (float)Lk;
    for (int l = 0; l < L; ++l) cx[(long)l * HD] = s;
  }
}

// Selected-query scores + softmax + weighted V, scattered into ctx
__global__ __launch_bounds__(128) void probupd_k(
    const float* __restrict__ qb, const float* __restrict__ kb,
    const float* __restrict__ vb, const int* __restrict__ top,
    float* __restrict__ ctx, int L, int Lk, int u, int maskf)
{
  __shared__ float s[512];
  __shared__ float red[128];
  int bh = blockIdx.x / u;
  int i  = blockIdx.x % u;
  int h = bh % NH, b = bh / NH;
  int tid = threadIdx.x;
  int qi = top[(long)bh * u + i];
  const float* q = qb + ((long)b * L + qi) * DM + h * HD;
  for (int k = tid; k < Lk; k += 128) {
    const float* kr = kb + ((long)b * Lk + k) * DM + h * HD;
    float d = 0.0f;
    for (int t = 0; t < HD; ++t) d += q[t] * kr[t];
    d *= 0.125f;                       // 1/sqrt(64)
    if (maskf && k > qi) d = -1e30f;
    s[k] = d;
  }
  __syncthreads();
  float mx = -1e30f;
  for (int k = tid; k < Lk; k += 128) mx = fmaxf(mx, s[k]);
  red[tid] = mx; __syncthreads();
  for (int o = 64; o > 0; o >>= 1) { if (tid < o) red[tid] = fmaxf(red[tid], red[tid + o]); __syncthreads(); }
  mx = red[0]; __syncthreads();
  float sum = 0.0f;
  for (int k = tid; k < Lk; k += 128) { float e = expf(s[k] - mx); s[k] = e; sum += e; }
  red[tid] = sum; __syncthreads();
  for (int o = 64; o > 0; o >>= 1) { if (tid < o) red[tid] += red[tid + o]; __syncthreads(); }
  float inv = 1.0f / red[0];
  __syncthreads();
  if (tid < HD) {
    const float* vc = vb + (long)b * Lk * DM + h * HD + tid;
    float a = 0.0f;
    for (int k = 0; k < Lk; ++k) a += s[k] * vc[(long)k * DM];
    ctx[((long)bh * L + qi) * HD + tid] = a * inv;
  }
}

// Head merge: mix=0 -> [b,l,h*64+d]; mix=1 -> raw [H,L,D] reshape to [L,512]
__global__ void merge_k(const float* __restrict__ ctx, float* __restrict__ ao,
                        int L, int mix)
{
  long idx = (long)blockIdx.x * blockDim.x + threadIdx.x;
  long tot = (long)BB * NH * L * HD;
  if (idx >= tot) return;
  int d = (int)(idx % HD);
  long r = idx / HD;
  int l = (int)(r % L);
  long bh = r / L;
  int h = (int)(bh % NH);
  int b = (int)(bh / NH);
  float v = ctx[idx];
  if (!mix) {
    ao[((long)b * L + l) * DM + h * HD + d] = v;
  } else {
    long flat = ((long)h * L + l) * HD + d;
    long lp = flat / DM;
    long cp = flat % DM;
    ao[((long)b * L + lp) * DM + cp] = v;
  }
}

// Final projection to C_OUT=7, last PRED_LEN=256 positions
__global__ void proj_k(const float* __restrict__ x, const float* __restrict__ w,
                       const float* __restrict__ b, float* __restrict__ out)
{
  int idx = blockIdx.x * blockDim.x + threadIdx.x;
  int tot = BB * 256 * 7;
  if (idx >= tot) return;
  int c = idx % 7;
  int r = idx / 7;
  int l = r % 256;
  int bb = r / 256;
  const float* xr = x + ((long)bb * 512 + 256 + l) * DM;
  float a = b[c];
  for (int d = 0; d < DM; ++d) a += xr[d] * w[d * 7 + c];
  out[idx] = a;
}

// ---------------------------------------------------------------------------
// Host orchestration
// ---------------------------------------------------------------------------
struct Lin  { const float *b, *w; };
struct LNp  { const float *b, *g; };
struct AttnP{ Lin k, o, q, v; };
struct EncP { AttnP attn; Lin ff1, ff2; LNp ln1, ln2; };
struct DecP { AttnP cross; Lin ff1, ff2; LNp ln1, ln2, ln3; AttnP self_; };
struct DistP{ const float *b, *bn_b, *bn_g, *w; };
struct EmbP { const float* conv_w; Lin temp; };

static int probU(int L) {
  int u = (int)(5.0 * ceil(log((double)L)) + 0.5);
  return u < L ? u : L;
}

extern "C" void kernel_launch(void* const* d_in, const int* in_sizes, int n_in,
                              void* d_out, int out_size, void* d_ws, size_t ws_size,
                              hipStream_t stream) {
  (void)in_sizes; (void)n_in; (void)out_size; (void)ws_size;

  const float* x_enc  = (const float*)d_in[0];
  const float* xm_enc = (const float*)d_in[1];
  const float* x_dec  = (const float*)d_in[2];
  const float* xm_dec = (const float*)d_in[3];

  // --- params: jax pytree leaves, dicts sorted by key ---
  int pi = 4;
  auto NX = [&]() { return (const float*)d_in[pi++]; };
  auto rdLin  = [&]() { Lin l;  l.b = NX(); l.w = NX(); return l; };
  auto rdLN   = [&]() { LNp l;  l.b = NX(); l.g = NX(); return l; };
  auto rdAttn = [&]() { AttnP a; a.k = rdLin(); a.o = rdLin(); a.q = rdLin(); a.v = rdLin(); return a; };

  DecP dec[2];
  for (int i = 0; i < 2; ++i) {
    dec[i].cross = rdAttn();
    dec[i].ff1 = rdLin(); dec[i].ff2 = rdLin();
    dec[i].ln1 = rdLN();  dec[i].ln2 = rdLN(); dec[i].ln3 = rdLN();
    dec[i].self_ = rdAttn();
  }
  EmbP dec_emb; dec_emb.conv_w = NX(); dec_emb.temp = rdLin();
  LNp dec_norm = rdLN();
  DistP dist[2];
  for (int i = 0; i < 2; ++i) {
    dist[i].b = NX(); dist[i].bn_b = NX(); dist[i].bn_g = NX(); dist[i].w = NX();
  }
  EncP enc[3];
  for (int i = 0; i < 3; ++i) {
    enc[i].attn = rdAttn();
    enc[i].ff1 = rdLin(); enc[i].ff2 = rdLin();
    enc[i].ln1 = rdLN();  enc[i].ln2 = rdLN();
  }
  EmbP enc_emb; enc_emb.conv_w = NX(); enc_emb.temp = rdLin();
  LNp enc_norm = rdLN();
  Lin proj = rdLin();

  // --- workspace layout (floats) ---
  float* ws = (float*)d_ws;
  const long SZ  = (long)BB * 512 * DM;   // 1,048,576
  const long SZP = (long)BB * 520 * DM;   // padded/conv scratch
  float* XP  = ws;
  float* T1  = XP  + SZP;
  float* X   = T1  + SZP;
  float* ENC = X   + SZ;
  float* Qb  = ENC + SZ;
  float* Kb  = Qb  + SZ;
  float* Vb  = Kb  + SZ;
  float* AO  = Vb  + SZ;
  float* T2  = AO  + SZ;
  float* SC  = T2  + SZ;                        // scores (<= 32*512*136)
  float* CTX = SC  + (long)BB * NH * 512 * 136;
  float* MM  = CTX + (long)BB * NH * 512 * HD;
  float* BN  = MM  + (long)BB * NH * 512;
  int*   TOP = (int*)(BN + 2 * DM);

  // simple (batch=1, lda=K, no transpose) GEMM
  auto G = [&](const float* A, const float* Wt, const float* bi, float* Cp,
               const float* rs, int M, int N, int K, int act) {
    dim3 grid((N + 63) / 64, (M + 127) / 128, 1);
    gemm_f16_wmma<0><<<grid, 128, 0, stream>>>(A, Wt, bi, Cp, rs, M, N, K, K, N, N,
                                               0, 0, 0, 0, 0, 0, 1, act, 1.0f);
  };

  // ================= Encoder =================
  int L = 512;
  embed_k<<<BB * L, DM, 0, stream>>>(x_enc, xm_enc, enc_emb.conv_w,
                                     enc_emb.temp.w, enc_emb.temp.b, X, L, 7);
  for (int li = 0; li < 3; ++li) {
    EncP& ep = enc[li];
    int M = BB * L;
    G(X, ep.attn.q.w, ep.attn.q.b, Qb, nullptr, M, DM, DM, 0);
    G(X, ep.attn.k.w, ep.attn.k.b, Kb, nullptr, M, DM, DM, 0);
    G(X, ep.attn.v.w, ep.attn.v.b, Vb, nullptr, M, DM, DM, 0);
    int U = probU(L);
    long qt = (long)BB * NH * L;
    probM_k<<<(int)((qt + 127) / 128), 128, 0, stream>>>(Qb, Kb, MM, L, L, U);
    topk_k<<<BB * NH, 256, 0, stream>>>(MM, TOP, L, U);
    ctxinit_k<<<BB * NH, 64, 0, stream>>>(Vb, CTX, L, L, 0);
    probupd_k<<<BB * NH * U, 128, 0, stream>>>(Qb, Kb, Vb, TOP, CTX, L, L, U, 0);
    long mt = (long)BB * NH * L * HD;
    merge_k<<<(int)((mt + 255) / 256), 256, 0, stream>>>(CTX, AO, L, 0);
    G(AO, ep.attn.o.w, ep.attn.o.b, T2, X, M, DM, DM, 0);
    layernorm_k<<<M, 256, 0, stream>>>(T2, X, ep.ln1.g, ep.ln1.b);
    G(X, ep.ff1.w, ep.ff1.b, Qb, nullptr, M, DM, DM, 1);
    G(Qb, ep.ff2.w, ep.ff2.b, T2, X, M, DM, DM, 0);
    layernorm_k<<<M, 256, 0, stream>>>(T2, X, ep.ln2.g, ep.ln2.b);

    if (li < 2) {
      DistP& dp = dist[li];
      long pt = (long)BB * (L + 4) * DM;
      padcirc_k<<<(int)((pt + 255) / 256), 256, 0, stream>>>(X, XP, L, 2);
      int Lc = L + 2;
      // conv width-3 as GEMM over K = 3*512 (3 contiguous padded rows)
      dim3 gc((DM + 63) / 64, (Lc + 127) / 128, BB);
      gemm_f16_wmma<0><<<gc, 128, 0, stream>>>(XP, dp.w, dp.b, T1, nullptr,
          Lc, DM, 3 * DM, DM, DM, DM,
          (long)(L + 4) * DM, 0, 0, 0, (long)Lc * DM, 0, 1, 0, 1.0f);
      bnstats_k<<<DM, 256, 0, stream>>>(T1, BN, BB * Lc);
      long et = (long)BB * Lc * DM;
      bnelu_k<<<(int)((et + 255) / 256), 256, 0, stream>>>(T1, BN, dp.bn_g, dp.bn_b, et);
      int Lo = (Lc - 1) / 2 + 1;
      long ot = (long)BB * Lo * DM;
      maxpool_k<<<(int)((ot + 255) / 256), 256, 0, stream>>>(T1, X, Lc, Lo);
      L = Lo;
    }
  }
  layernorm_k<<<BB * L, 256, 0, stream>>>(X, ENC, enc_norm.g, enc_norm.b);
  int Le = L;

  // ================= Decoder =================
  int Ld = 512, Md = BB * Ld;
  embed_k<<<BB * Ld, DM, 0, stream>>>(x_dec, xm_dec, dec_emb.conv_w,
                                      dec_emb.temp.w, dec_emb.temp.b, X, Ld, 7);
  for (int li = 0; li < 2; ++li) {
    DecP& dp = dec[li];
    // --- masked prob self-attention (mix=True) ---
    G(X, dp.self_.q.w, dp.self_.q.b, Qb, nullptr, Md, DM, DM, 0);
    G(X, dp.self_.k.w, dp.self_.k.b, Kb, nullptr, Md, DM, DM, 0);
    G(X, dp.self_.v.w, dp.self_.v.b, Vb, nullptr, Md, DM, DM, 0);
    int U = probU(Ld);
    long qt = (long)BB * NH * Ld;
    probM_k<<<(int)((qt + 127) / 128), 128, 0, stream>>>(Qb, Kb, MM, Ld, Ld, U);
    topk_k<<<BB * NH, 256, 0, stream>>>(MM, TOP, Ld, U);
    ctxinit_k<<<BB * NH, 64, 0, stream>>>(Vb, CTX, Ld, Ld, 1);
    probupd_k<<<BB * NH * U, 128, 0, stream>>>(Qb, Kb, Vb, TOP, CTX, Ld, Ld, U, 1);
    long mt = (long)BB * NH * Ld * HD;
    merge_k<<<(int)((mt + 255) / 256), 256, 0, stream>>>(CTX, AO, Ld, 1);
    G(AO, dp.self_.o.w, dp.self_.o.b, T2, X, Md, DM, DM, 0);
    layernorm_k<<<Md, 256, 0, stream>>>(T2, X, dp.ln1.g, dp.ln1.b);

    // --- full cross-attention against encoder output ---
    G(X,   dp.cross.q.w, dp.cross.q.b, Qb, nullptr, Md,      DM, DM, 0);
    G(ENC, dp.cross.k.w, dp.cross.k.b, Kb, nullptr, BB * Le, DM, DM, 0);
    G(ENC, dp.cross.v.w, dp.cross.v.b, Vb, nullptr, BB * Le, DM, DM, 0);
    // scores[z=(b,h)] = Q[b,:,h] @ K[b,:,h]^T * 1/8   (batched, transB)
    {
      dim3 gs((Le + 63) / 64, (Ld + 127) / 128, BB * NH);
      gemm_f16_wmma<1><<<gs, 128, 0, stream>>>(Qb, Kb, nullptr, SC, nullptr,
          Ld, Le, HD, DM, DM, Le,
          (long)Ld * DM, HD, (long)Le * DM, HD,
          (long)NH * Ld * Le, (long)Ld * Le,
          NH, 0, 0.125f);
    }
    softmax_k<<<BB * NH * Ld, 128, 0, stream>>>(SC, Le);
    // AO[b,l,h*64+d] = P @ V  (batched)
    {
      dim3 gp((HD + 63) / 64, (Ld + 127) / 128, BB * NH);
      gemm_f16_wmma<0><<<gp, 128, 0, stream>>>(SC, Vb, nullptr, AO, nullptr,
          Ld, HD, Le, Le, DM, DM,
          (long)NH * Ld * Le, (long)Ld * Le,
          (long)Le * DM, HD,
          (long)Ld * DM, HD,
          NH, 0, 1.0f);
    }
    G(AO, dp.cross.o.w, dp.cross.o.b, T2, X, Md, DM, DM, 0);
    layernorm_k<<<Md, 256, 0, stream>>>(T2, X, dp.ln2.g, dp.ln2.b);

    // --- FFN ---
    G(X, dp.ff1.w, dp.ff1.b, Qb, nullptr, Md, DM, DM, 1);
    G(Qb, dp.ff2.w, dp.ff2.b, T2, X, Md, DM, DM, 0);
    layernorm_k<<<Md, 256, 0, stream>>>(T2, X, dp.ln3.g, dp.ln3.b);
  }
  layernorm_k<<<Md, 256, 0, stream>>>(X, T2, dec_norm.g, dec_norm.b);
  proj_k<<<(BB * 256 * 7 + 127) / 128, 128, 0, stream>>>(T2, proj.w, proj.b,
                                                         (float*)d_out);
}
